// FDSA_24962349924671
// MI455X (gfx1250) — matrix operand order repaired and checked by reading, not compiled, not run on hardware
//
#include <hip/hip_runtime.h>
#include <hip/hip_bf16.h>
#include <math.h>

// ---------------------------------------------------------------------------
// FDSA block for MI455X (gfx1250, wave32, WMMA).
// Sizes: B=2, DIM=64, E=76, E4=304, H=W=256, P=8.
// N = B*H*W = 131072 pixels.  HW = 65536.
// Pipeline:
//   1. cvt_x      : x (f32 [B,64,H,W]) -> xT f16 [N][64]   (pixel-major for WMMA B-frag)
//   2. cvt_whid   : w_hidden f32 [304][64] -> f16
//   3. cvt_wproj  : w_proj f32 [64][228] -> f16 [64][256] (zero-padded K)
//   4. gemm1      : hidden[304][N] = Wh x xT^T   (v_wmma_f32_16x16x32_f16), store f16
//   5. dwconv     : depthwise 3x3 SAME, f16 in / f32 out  hdw[304][N]
//   6. fft        : per (c,patch): rfft2(8x8) on q,k,v; phase/mag mixing; irfft2 -> O1/O2/O3
//   7. stats      : layernorm mean / rstd over 76 channels per pixel (x3)
//   8. normcat    : LN + v_value multiply + pack catT f16 [N][256] (cols 228..255 = 0)
//   9. gemm2      : d_out[64][N] = Wp x catT^T  (WMMA, K=256), scatter to NCHW f32
// Workspace requirement ~446 MB.
// ---------------------------------------------------------------------------

#define NTOT   131072   // B*H*W
#define HWSZ   65536
#define ECH    76
#define E4CH   304
#define DIMCH  64
#define KPAD   256      // padded 3*E = 228 -> 256

typedef _Float16 v8h  __attribute__((ext_vector_type(8)));
typedef _Float16 v16h __attribute__((ext_vector_type(16)));
typedef float    v8f  __attribute__((ext_vector_type(8)));

__device__ __forceinline__ v16h cat8(v8h lo, v8h hi) {
  return __builtin_shufflevector(lo, hi, 0,1,2,3,4,5,6,7,8,9,10,11,12,13,14,15);
}

// One 16x16 output tile, A[M][K] row-major f16, Bt[Nc][K] row-major f16 (B transposed).
// ISA layouts: A 16x32 f16 -> lane half kg=(l>>4): K = kg*8+{0..7} then 16+kg*8+{0..7}.
//              B 32x16 f16 -> lane half kg: K = kg*8+{0..7} then 16+kg*8+{0..7}.
template<int K>
__device__ __forceinline__ v8f wmma_tile16(const _Float16* __restrict__ A,
                                           const _Float16* __restrict__ Bt,
                                           int rowA, int colB, int lane) {
  const int kg = lane >> 4;
  v8f acc = {};
  const _Float16* arow = A  + (size_t)rowA * K;
  const _Float16* brow = Bt + (size_t)colB * K;
#pragma unroll
  for (int kb = 0; kb < K; kb += 32) {
    if (kb + 32 < K) {                       // prefetch next K-block of B
      __builtin_prefetch(brow + kb + 32, 0, 1);
    }
    const _Float16* ap = arow + kb + kg * 8;
    v8h a_lo = *(const v8h*)(ap);
    v8h a_hi = *(const v8h*)(ap + 16);
    const _Float16* bp = brow + kb + kg * 8;
    v8h b_lo = *(const v8h*)(bp);
    v8h b_hi = *(const v8h*)(bp + 16);
    v16h a = cat8(a_lo, a_hi);
    v16h b = cat8(b_lo, b_hi);
    acc = __builtin_amdgcn_wmma_f32_16x16x32_f16(
        /*neg_a=*/false, a, /*neg_b=*/false, b,
        /*c_mod=*/(short)0, acc, /*reuse_a=*/false, /*reuse_b=*/false);
  }
  return acc;
}

// ------------------------- conversion kernels ------------------------------

__global__ void k_cvt_x(const float* __restrict__ x, _Float16* __restrict__ xT) {
  size_t id = (size_t)blockIdx.x * blockDim.x + threadIdx.x;  // NTOT*64
  if (id >= (size_t)NTOT * 64) return;
  int c = (int)(id & 63);
  int n = (int)(id >> 6);
  int b  = n >> 16;
  int hw = n & (HWSZ - 1);
  xT[id] = (_Float16)x[(size_t)b * 64 * HWSZ + (size_t)c * HWSZ + hw];
}

__global__ void k_cvt_whid(const float* __restrict__ w, _Float16* __restrict__ wh) {
  int id = blockIdx.x * blockDim.x + threadIdx.x;   // 304*64
  if (id >= E4CH * DIMCH) return;
  wh[id] = (_Float16)w[id];
}

__global__ void k_cvt_wproj(const float* __restrict__ w, _Float16* __restrict__ wp) {
  int id = blockIdx.x * blockDim.x + threadIdx.x;   // 64*256
  if (id >= DIMCH * KPAD) return;
  int o = id >> 8, c = id & 255;
  wp[id] = (c < 3 * ECH) ? (_Float16)w[o * (3 * ECH) + c] : (_Float16)0.0f;
}

// ------------------------------ GEMM 1 -------------------------------------
// hidden[m][n] = sum_c Wh[m][c] * xT[n][c] ; M=304, K=64, N=131072.
__global__ void k_gemm1(const _Float16* __restrict__ Wh,
                        const _Float16* __restrict__ xT,
                        _Float16* __restrict__ Hid) {
  const int lane  = threadIdx.x & 31;
  const int wave  = threadIdx.x >> 5;
  const int tileN = blockIdx.x * 8 + wave;      // 1024*8 = 8192 tiles
  const int tileM = blockIdx.y;                 // 19 tiles
  const int row = tileM * 16 + (lane & 15);
  const int col = tileN * 16 + (lane & 15);
  v8f acc = wmma_tile16<DIMCH>(Wh, xT, row, col, lane);
  const int mbase = tileM * 16 + ((lane >> 4) << 3);
#pragma unroll
  for (int r = 0; r < 8; ++r)
    Hid[(size_t)(mbase + r) * NTOT + col] = (_Float16)acc[r];
}

// --------------------------- depthwise 3x3 ---------------------------------
__global__ void k_dwconv(const _Float16* __restrict__ Hid,
                         const float* __restrict__ Wdw,
                         float* __restrict__ Hdw) {
  size_t id = (size_t)blockIdx.x * blockDim.x + threadIdx.x;  // 304*NTOT
  if (id >= (size_t)E4CH * NTOT) return;
  int n = (int)(id & (NTOT - 1));
  int c = (int)(id >> 17);                  // NTOT = 2^17
  int b  = n >> 16;
  int hw = n & (HWSZ - 1);
  int h = hw >> 8, w = hw & 255;
  const _Float16* base = Hid + (size_t)c * NTOT + (size_t)b * HWSZ;
  const float* wk = Wdw + c * 9;
  float s = 0.0f;
#pragma unroll
  for (int dy = -1; dy <= 1; ++dy) {
    int hh = h + dy;
    if (hh < 0 || hh > 255) continue;
#pragma unroll
    for (int dx = -1; dx <= 1; ++dx) {
      int ww = w + dx;
      if (ww < 0 || ww > 255) continue;
      s += (float)base[hh * 256 + ww] * wk[(dy + 1) * 3 + (dx + 1)];
    }
  }
  Hdw[id] = s;
}

// ------------------------- 8x8 DFT helpers ---------------------------------
__device__ const float COS8[8] = { 1.0f,  0.70710678f, 0.0f, -0.70710678f,
                                  -1.0f, -0.70710678f, 0.0f,  0.70710678f };
__device__ const float SIN8[8] = { 0.0f,  0.70710678f, 1.0f,  0.70710678f,
                                   0.0f, -0.70710678f,-1.0f, -0.70710678f };
#define THRD 1e-10f
__device__ __forceinline__ float rdn(float x) {
  return (x < THRD && x > -THRD) ? THRD : x;
}

__device__ void rfft2_8x8(const float (*p)[8], float (*Xr)[5], float (*Xi)[5]) {
  float rr[8][5], ri[8][5];
  for (int y = 0; y < 8; ++y)
    for (int v = 0; v < 5; ++v) {
      float sr = 0.0f, si = 0.0f;
#pragma unroll
      for (int x = 0; x < 8; ++x) {
        int t = (v * x) & 7;
        sr += p[y][x] * COS8[t];
        si -= p[y][x] * SIN8[t];
      }
      rr[y][v] = sr; ri[y][v] = si;
    }
  for (int u = 0; u < 8; ++u)
    for (int v = 0; v < 5; ++v) {
      float sr = 0.0f, si = 0.0f;
#pragma unroll
      for (int y = 0; y < 8; ++y) {
        int t = (u * y) & 7;
        float cc = COS8[t], ss = SIN8[t];
        sr += rr[y][v] * cc + ri[y][v] * ss;
        si += ri[y][v] * cc - rr[y][v] * ss;
      }
      Xr[u][v] = sr; Xi[u][v] = si;
    }
}

__device__ void irfft2_8x8(const float (*Sr)[5], const float (*Si)[5], float (*out)[8]) {
  float tr[8][5], ti[8][5];
  for (int y = 0; y < 8; ++y)
    for (int v = 0; v < 5; ++v) {
      float sr = 0.0f, si = 0.0f;
#pragma unroll
      for (int u = 0; u < 8; ++u) {
        int t = (u * y) & 7;
        float cc = COS8[t], ss = SIN8[t];
        sr += Sr[u][v] * cc - Si[u][v] * ss;
        si += Si[u][v] * cc + Sr[u][v] * ss;
      }
      tr[y][v] = sr; ti[y][v] = si;
    }
  const float WV[5] = {1.0f, 2.0f, 2.0f, 2.0f, 1.0f};
  for (int y = 0; y < 8; ++y)
    for (int x = 0; x < 8; ++x) {
      float s = 0.0f;
#pragma unroll
      for (int v = 0; v < 5; ++v) {
        int t = (v * x) & 7;
        s += WV[v] * (tr[y][v] * COS8[t] - ti[y][v] * SIN8[t]);
      }
      out[y][x] = s * (1.0f / 64.0f);
    }
}

// ---------------------- spectral attention stage ---------------------------
__global__ void k_fft(const float* __restrict__ Hdw,
                      const float* __restrict__ fftp,
                      float* __restrict__ O123) {
  int id = blockIdx.x * blockDim.x + threadIdx.x;   // 76 * 2048 = 155648
  if (id >= ECH * 2048) return;
  int patch = id & 2047;         // b*1024 + ph*32 + pw
  int c     = id >> 11;
  int b  = patch >> 10;
  int ph = (patch >> 5) & 31;
  int pw = patch & 31;
  size_t pixbase = (size_t)b * HWSZ + (size_t)(ph * 8) * 256 + (size_t)(pw * 8);

  float p[8][8];
  float qka[8][5], cqk[8][5], sqk[8][5];
  {
    float qr[8][5], qi[8][5], kr[8][5], ki[8][5];
    {
      const float* src = Hdw + (size_t)c * NTOT + pixbase;
      for (int y = 0; y < 8; ++y)
        for (int x = 0; x < 8; ++x) p[y][x] = src[y * 256 + x];
      rfft2_8x8(p, qr, qi);
    }
    {
      const float* src = Hdw + (size_t)(ECH + c) * NTOT + pixbase;
      for (int y = 0; y < 8; ++y)
        for (int x = 0; x < 8; ++x) p[y][x] = src[y * 256 + x];
      rfft2_8x8(p, kr, ki);
    }
    for (int u = 0; u < 8; ++u)
      for (int v = 0; v < 5; ++v) {
        float ar = qr[u][v], ai = qi[u][v];
        float br = kr[u][v], bi = ki[u][v];
        float pr = rdn(ar * br - ai * bi);
        float pi = rdn(ar * bi + ai * br);
        qka[u][v] = sqrtf(pr * pr + pi * pi);
        float pa = atan2f(rdn(ai), rdn(ar)) - atan2f(rdn(bi), rdn(br));
        cqk[u][v] = cosf(pa);
        sqk[u][v] = sinf(pa);
      }
  }
  float s1r[8][5], s1i[8][5], s2r[8][5], s2i[8][5], s3r[8][5], s3i[8][5];
  {
    float vr[8][5], vi[8][5];
    const float* src = Hdw + (size_t)(2 * ECH + c) * NTOT + pixbase;
    for (int y = 0; y < 8; ++y)
      for (int x = 0; x < 8; ++x) p[y][x] = src[y * 256 + x];
    rfft2_8x8(p, vr, vi);
    for (int u = 0; u < 8; ++u)
      for (int v = 0; v < 5; ++v) {
        float f  = fftp[c * 40 + u * 5 + v];
        float ar = rdn(vr[u][v] * f);
        float ai = rdn(vi[u][v] * f);
        float va = sqrtf(ar * ar + ai * ai);
        float vp = atan2f(ai, ar);
        s1r[u][v] = va * cqk[u][v];        s1i[u][v] = va * sqk[u][v];
        s2r[u][v] = qka[u][v] * cosf(vp);  s2i[u][v] = qka[u][v] * sinf(vp);
        s3r[u][v] = qka[u][v] * cqk[u][v]; s3i[u][v] = qka[u][v] * sqk[u][v];
      }
  }
  float o[8][8];
  irfft2_8x8(s1r, s1i, o);
  {
    float* dst = O123 + (size_t)c * NTOT + pixbase;
    for (int y = 0; y < 8; ++y)
      for (int x = 0; x < 8; ++x) dst[y * 256 + x] = o[y][x];
  }
  irfft2_8x8(s2r, s2i, o);
  {
    float* dst = O123 + (size_t)(ECH + c) * NTOT + pixbase;
    for (int y = 0; y < 8; ++y)
      for (int x = 0; x < 8; ++x) dst[y * 256 + x] = o[y][x];
  }
  irfft2_8x8(s3r, s3i, o);
  {
    float* dst = O123 + (size_t)(2 * ECH + c) * NTOT + pixbase;
    for (int y = 0; y < 8; ++y)
      for (int x = 0; x < 8; ++x) dst[y * 256 + x] = o[y][x];
  }
}

// ------------------------- layernorm statistics ----------------------------
__global__ void k_stats(const float* __restrict__ O123,
                        float* __restrict__ Mu, float* __restrict__ Rstd) {
  int n = blockIdx.x * blockDim.x + threadIdx.x;
  if (n >= NTOT) return;
  for (int j = 0; j < 3; ++j) {
    const float* t = O123 + (size_t)j * ECH * NTOT + n;
    float mu = 0.0f;
    for (int c = 0; c < ECH; ++c) mu += t[(size_t)c * NTOT];
    mu *= (1.0f / (float)ECH);
    float var = 0.0f;
    for (int c = 0; c < ECH; ++c) {
      float d = t[(size_t)c * NTOT] - mu;
      var += d * d;
    }
    var *= (1.0f / (float)ECH);
    Mu  [j * NTOT + n] = mu;
    Rstd[j * NTOT + n] = rsqrtf(var + 1e-5f);
  }
}

// -------------------- LN apply + v_value mul + pack ------------------------
__global__ void k_normcat(const float* __restrict__ O123,
                          const float* __restrict__ Hdw,
                          const float* __restrict__ Mu,
                          const float* __restrict__ Rstd,
                          const float* __restrict__ ln1w, const float* __restrict__ ln1b,
                          const float* __restrict__ ln2w, const float* __restrict__ ln2b,
                          const float* __restrict__ ln3w, const float* __restrict__ ln3b,
                          _Float16* __restrict__ Ct) {
  size_t id = (size_t)blockIdx.x * blockDim.x + threadIdx.x;  // NTOT*256
  if (id >= (size_t)NTOT * KPAD) return;
  int ccol = (int)(id & 255);
  int n    = (int)(id >> 8);
  if (ccol >= 3 * ECH) { Ct[id] = (_Float16)0.0f; return; }
  int j  = ccol / ECH;
  int cc = ccol - j * ECH;
  const float* lw = (j == 0) ? ln1w : (j == 1) ? ln2w : ln3w;
  const float* lb = (j == 0) ? ln1b : (j == 1) ? ln2b : ln3b;
  float val = O123[((size_t)j * ECH + cc) * NTOT + n];
  float mu  = Mu  [j * NTOT + n];
  float rs  = Rstd[j * NTOT + n];
  val = (val - mu) * rs * lw[cc] + lb[cc];
  float vv = Hdw[(size_t)(3 * ECH + cc) * NTOT + n];   // v_value channels
  Ct[id] = (_Float16)(val * vv);
}

// ------------------------------ GEMM 2 -------------------------------------
// out[o][n] = sum_c Wp[o][c] * Ct[n][c] ; M=64, K=256, N=131072; NCHW scatter.
__global__ void k_gemm2(const _Float16* __restrict__ Wp,
                        const _Float16* __restrict__ Ct,
                        float* __restrict__ out) {
  const int lane  = threadIdx.x & 31;
  const int wave  = threadIdx.x >> 5;
  const int tileN = blockIdx.x * 8 + wave;
  const int tileM = blockIdx.y;                 // 4 tiles of M=64
  const int row = tileM * 16 + (lane & 15);
  const int col = tileN * 16 + (lane & 15);
  v8f acc = wmma_tile16<KPAD>(Wp, Ct, row, col, lane);
  const int mbase = tileM * 16 + ((lane >> 4) << 3);
  const int b  = col >> 16;
  const int hw = col & (HWSZ - 1);
#pragma unroll
  for (int r = 0; r < 8; ++r)
    out[((size_t)(b * DIMCH + mbase + r)) * HWSZ + hw] = acc[r];
}

// ------------------------------ launcher -----------------------------------
extern "C" void kernel_launch(void* const* d_in, const int* in_sizes, int n_in,
                              void* d_out, int out_size, void* d_ws, size_t ws_size,
                              hipStream_t stream) {
  const float* x        = (const float*)d_in[0];
  const float* w_hidden = (const float*)d_in[1];
  const float* w_dw     = (const float*)d_in[2];
  const float* w_proj   = (const float*)d_in[3];
  const float* fft_p    = (const float*)d_in[4];
  const float* ln1w = (const float*)d_in[5];  const float* ln1b = (const float*)d_in[6];
  const float* ln2w = (const float*)d_in[7];  const float* ln2b = (const float*)d_in[8];
  const float* ln3w = (const float*)d_in[9];  const float* ln3b = (const float*)d_in[10];
  float* out = (float*)d_out;

  // ---- workspace bump allocator (256B aligned); ~446 MB total ----
  char* wp_ = (char*)d_ws;
  auto alloc = [&](size_t bytes) -> char* {
    char* p = wp_;
    wp_ += (bytes + 255) & ~(size_t)255;
    return p;
  };
  _Float16* xT    = (_Float16*)alloc((size_t)NTOT * 64 * 2);        //  16.8 MB
  _Float16* wh16  = (_Float16*)alloc((size_t)E4CH * DIMCH * 2);     //  39 KB
  _Float16* wp16  = (_Float16*)alloc((size_t)DIMCH * KPAD * 2);     //  33 KB
  _Float16* Hid   = (_Float16*)alloc((size_t)E4CH * NTOT * 2);      //  79.7 MB
  float*    Hdw   = (float*)   alloc((size_t)E4CH * NTOT * 4);      // 159.4 MB
  float*    O123  = (float*)   alloc((size_t)3 * ECH * NTOT * 4);   // 119.5 MB
  float*    Mu    = (float*)   alloc((size_t)3 * NTOT * 4);         //   1.6 MB
  float*    Rstd  = (float*)   alloc((size_t)3 * NTOT * 4);         //   1.6 MB
  _Float16* Ct    = (_Float16*)alloc((size_t)NTOT * KPAD * 2);      //  67.1 MB
  (void)ws_size; (void)in_sizes; (void)n_in; (void)out_size;

  const int T = 256;
  // 1-3: conversions
  k_cvt_x    <<<(unsigned)(((size_t)NTOT * 64 + T - 1) / T), T, 0, stream>>>(x, xT);
  k_cvt_whid <<<(E4CH * DIMCH + T - 1) / T, T, 0, stream>>>(w_hidden, wh16);
  k_cvt_wproj<<<(DIMCH * KPAD + T - 1) / T, T, 0, stream>>>(w_proj, wp16);
  // 4: hidden GEMM  (8192 N-tiles / 8 waves, 19 M-tiles)
  k_gemm1<<<dim3(NTOT / 16 / 8, E4CH / 16), T, 0, stream>>>(wh16, xT, Hid);
  // 5: depthwise 3x3
  k_dwconv<<<(unsigned)(((size_t)E4CH * NTOT + T - 1) / T), T, 0, stream>>>(Hid, w_dw, Hdw);
  // 6: spectral stage (76 channels x 2048 patches)
  k_fft<<<(ECH * 2048 + T - 1) / T, T, 0, stream>>>(Hdw, fft_p, O123);
  // 7-8: layernorm + pack
  k_stats<<<(NTOT + T - 1) / T, T, 0, stream>>>(O123, Mu, Rstd);
  k_normcat<<<(unsigned)(((size_t)NTOT * KPAD + T - 1) / T), T, 0, stream>>>(
      O123, Hdw, Mu, Rstd, ln1w, ln1b, ln2w, ln2b, ln3w, ln3b, Ct);
  // 9: projection GEMM
  k_gemm2<<<dim3(NTOT / 16 / 8, DIMCH / 16), T, 0, stream>>>(wp16, Ct, out);
}
